// BidirLSTMLayer_88115549045326
// MI455X (gfx1250) — matrix operand order repaired and checked by reading, not compiled, not run on hardware
//
#include <hip/hip_runtime.h>
#include <hip/hip_bf16.h>

#define SEQ 512
#define BSZ 64
#define INP 512
#define HID 512
#define G4  (4 * HID)   // 2048 gate columns

typedef __bf16 bf16_t;
typedef __attribute__((ext_vector_type(16))) __bf16 v16bf;
typedef __attribute__((ext_vector_type(8)))  __bf16 v8bf;
typedef __attribute__((ext_vector_type(8)))  float  v8f;

// ---------------------------------------------------------------------------
// Fast activations (v_exp_f32 / v_rcp_f32)
// ---------------------------------------------------------------------------
__device__ __forceinline__ float fast_sigmoid(float x) {
    return __frcp_rn(1.0f + __expf(-x));
}
__device__ __forceinline__ float fast_tanh(float x) {
    float e = __expf(2.0f * x);
    return (e - 1.0f) * __frcp_rn(e + 1.0f);
}

// ---------------------------------------------------------------------------
// WMMA tile loaders (wave32); A and B both use this code's contiguous k order k = 16h + s.
//
// A-matrix 16x32 bf16, row-major A[m, k], leading dim lda:
//   lane L: m = m0 + (L & 15); kh = (L<16) ? 0 : 16
//   elems 0..7  = A[m, k0+kh+0 .. +7]
//   elems 8..15 = A[m, k0+kh+8 .. +15]
// ---------------------------------------------------------------------------
__device__ __forceinline__ v16bf load_a_tile(const bf16_t* __restrict__ A,
                                             int lda, int m0, int k0, int lane) {
    const int m  = m0 + (lane & 15);
    const int kh = (lane < 16) ? 0 : 16;
    const bf16_t* p = A + (size_t)m * lda + (k0 + kh);
    v8bf lo = *(const v8bf*)(p);
    v8bf hi = *(const v8bf*)(p + 8);
    v16bf r;
#pragma unroll
    for (int i = 0; i < 8; ++i) { r[i] = lo[i]; r[i + 8] = hi[i]; }
    return r;
}

// B-matrix 32x16 (KxN) with B[k,n] = W[n,k], W row-major [N, K] leading dim ldk:
//   lane L: n = n0 + (L & 15); ko = (L<16) ? 0 : 16
//   elems e = W[n, k0+ko+e], e=0..15  (one contiguous 32-byte load)
__device__ __forceinline__ v16bf load_b_tile(const bf16_t* __restrict__ W,
                                             int ldk, int n0, int k0, int lane) {
    const int n  = n0 + (lane & 15);
    const int ko = (lane < 16) ? 0 : 16;
    return *(const v16bf*)(W + (size_t)n * ldk + (k0 + ko));
}

__device__ __forceinline__ v8f wmma_bf16(v16bf a, v16bf b, v8f c) {
    return __builtin_amdgcn_wmma_f32_16x16x32_bf16(
        /*neg_a=*/false, a, /*neg_b=*/false, b,
        /*c_mod=*/(short)0, c, /*reuse_a=*/false, /*reuse_b=*/false);
}

// ---------------------------------------------------------------------------
// Small init kernels
// ---------------------------------------------------------------------------
__global__ void k_cvt_f32_to_bf16(bf16_t* __restrict__ dst,
                                  const float* __restrict__ src, int n) {
    int i = blockIdx.x * blockDim.x + threadIdx.x;
    int stride = gridDim.x * blockDim.x;
    for (; i < n; i += stride) dst[i] = (bf16_t)src[i];
}

__global__ void k_bias_sum(float* __restrict__ dst,
                           const float* __restrict__ a,
                           const float* __restrict__ b, int n) {
    int i = blockIdx.x * blockDim.x + threadIdx.x;
    if (i < n) dst[i] = a[i] + b[i];
}

__global__ void k_zero_u32(unsigned* p) { *p = 0u; }

// ---------------------------------------------------------------------------
// Phase 2: time-parallel input GEMM
//   Gpre[dir, m=t*B+b, n] = Xbf[m,:] @ Wih[dir]^T + (bih+bhh)[dir][n]
// Block = 128 threads (4 waves). wave w -> Mtile = blockIdx.x*4 + w.
// Each wave computes 16 rows x 64 cols (4 N-tiles sharing the A tile).
// Grid: (TB/64, G4/64, 2) = (512, 32, 2).
// ---------------------------------------------------------------------------
__global__ void __launch_bounds__(128)
k_input_gemm(const bf16_t* __restrict__ Xbf,    // [T*B][INP]
             const bf16_t* __restrict__ WihBf,  // [2][G4][INP]
             const float*  __restrict__ bsum,   // [2][G4]
             float*        __restrict__ Gpre) { // [2][T*B][G4]
    const int dir   = blockIdx.z;
    const int wave  = threadIdx.x >> 5;
    const int lane  = threadIdx.x & 31;
    const int m0    = (blockIdx.x * 4 + wave) * 16;
    const int nbase = blockIdx.y * 64;

    const bf16_t* W  = WihBf + (size_t)dir * G4 * INP;
    const float*  bs = bsum  + (size_t)dir * G4;
    float*        G  = Gpre  + (size_t)dir * SEQ * BSZ * G4;

    const int n  = lane & 15;
    const int mb = (lane < 16) ? 0 : 8;

    v8f acc[4];
#pragma unroll
    for (int g = 0; g < 4; ++g) {
        float bv = bs[nbase + g * 16 + n];
#pragma unroll
        for (int r = 0; r < 8; ++r) acc[g][r] = bv;
    }

    for (int k0 = 0; k0 < INP; k0 += 32) {
        v16bf a = load_a_tile(Xbf, INP, m0, k0, lane);
#pragma unroll
        for (int g = 0; g < 4; ++g) {
            v16bf b = load_b_tile(W, INP, nbase + g * 16, k0, lane);
            acc[g] = wmma_bf16(a, b, acc[g]);
        }
    }

#pragma unroll
    for (int g = 0; g < 4; ++g) {
        const int col = nbase + g * 16 + n;
#pragma unroll
        for (int r = 0; r < 8; ++r)
            G[(size_t)(m0 + mb + r) * G4 + col] = acc[g][r];
    }
}

// ---------------------------------------------------------------------------
// Phase 3: persistent recurrence kernel. One launch total.
//
// Grid: (HID/16, 2) = (32, 2); block = 128 threads (4 waves, tileM = wave).
// Each block owns a 16-hidden-unit column slab for one direction and keeps
// the four Whh gate fragment tiles (4 x K=512 x N=16 bf16 = 64 KB) in LDS in
// WMMA per-lane fragment layout, so every step's 64 WMMAs read B operands
// via ds_load instead of global.
//
// Step-to-step sync: device-wide software barrier on a monotonically
// increasing counter (agent-scope release add / acquire spin + s_sleep).
// h-state ping-pongs on step parity; c is owner-exclusive per block.
// ---------------------------------------------------------------------------
#define NBLOCKS (2 * (HID / 16))   // 64 blocks total

__global__ void __launch_bounds__(128)
k_lstm_persistent(const float*  __restrict__ Gpre,    // [2][T*B][G4]
                  const bf16_t* __restrict__ WhhBf,   // [2][G4][HID]
                  bf16_t*       __restrict__ h_state, // [2 dir][2 ping][B][H]
                  float*        __restrict__ c_state, // [2 dir][B][H]
                  float*        __restrict__ out,     // [T][B][2H]
                  float*        __restrict__ finals,  // hT_f,cT_f,hT_b,cT_b
                  unsigned*     __restrict__ barrier_cnt) {
    // B fragments: [gate][kchunk][lane] -> 16 bf16 (32 B) each = 64 KB
    __shared__ __align__(32) bf16_t bsh[4 * 16 * 32 * 16];

    const int dir  = blockIdx.y;
    const int u0   = blockIdx.x * 16;         // hidden-unit slab
    const int wave = threadIdx.x >> 5;
    const int lane = threadIdx.x & 31;
    const int m0   = wave * 16;               // batch rows for this wave

    const bf16_t* W = WhhBf + (size_t)dir * G4 * HID;
    float*        c = c_state + (size_t)dir * BSZ * HID;

    // ---- Prologue: stage Whh fragments into LDS (wave w loads gate w) ----
    {
        const int g = wave;
#pragma unroll
        for (int kc = 0; kc < 16; ++kc) {
            v16bf b = load_b_tile(W, HID, g * HID + u0, kc * 32, lane);
            *(v16bf*)&bsh[(((g << 4) + kc) * 32 + lane) << 4] = b;
        }
    }
    __syncthreads();

    const int n  = lane & 15;
    const int mb = (lane < 16) ? 0 : 8;
    const int u  = u0 + n;

    for (int s = 0; s < SEQ; ++s) {
        const int t = dir ? (SEQ - 1 - s) : s;
        const bf16_t* hin  = h_state + ((size_t)dir * 2 + (s & 1)) * BSZ * HID;
        bf16_t*       hout = h_state + ((size_t)dir * 2 + ((s + 1) & 1)) * BSZ * HID;
        const float*  G    = Gpre + ((size_t)dir * SEQ * BSZ + (size_t)t * BSZ) * G4;

        // Accumulators = precomputed input projection + biases.
        v8f acc[4];
#pragma unroll
        for (int g = 0; g < 4; ++g) {
            const int col = g * HID + u0 + n;
#pragma unroll
            for (int r = 0; r < 8; ++r)
                acc[g][r] = G[(size_t)(m0 + mb + r) * G4 + col];
        }

        // gates += h_prev @ Whh^T : A from global (L2-hot), B from LDS.
#pragma unroll 4
        for (int kc = 0; kc < 16; ++kc) {
            v16bf a = load_a_tile(hin, HID, m0, kc * 32, lane);
#pragma unroll
            for (int g = 0; g < 4; ++g) {
                v16bf b = *(const v16bf*)&bsh[(((g << 4) + kc) * 32 + lane) << 4];
                acc[g] = wmma_bf16(a, b, acc[g]);
            }
        }

        // Prefetch next step's Gpre rows (2 prefetches/lane cover the
        // 64 cachelines this wave will gather next iteration).
        if (s + 1 < SEQ) {
            const int tn = dir ? (SEQ - 2 - s) : (s + 1);
            const float* Gn = Gpre + ((size_t)dir * SEQ * BSZ + (size_t)tn * BSZ) * G4;
            const int pr = m0 + (lane & 15);
            const int pg = lane >> 4;
            __builtin_prefetch(&Gn[(size_t)pr * G4 + pg * HID + u0], 0, 3);
            __builtin_prefetch(&Gn[(size_t)pr * G4 + (pg + 2) * HID + u0], 0, 3);
        }

        // LSTM cell update (wave-local: all four gates live in acc[]).
#pragma unroll
        for (int r = 0; r < 8; ++r) {
            const int bb = m0 + mb + r;
            const size_t idx = (size_t)bb * HID + u;
            float ig = fast_sigmoid(acc[0][r]);
            float fg = fast_sigmoid(acc[1][r]);
            float gg = fast_tanh(acc[2][r]);
            float og = fast_sigmoid(acc[3][r]);
            float cv = fg * c[idx] + ig * gg;
            c[idx] = cv;
            float hv = og * fast_tanh(cv);
            hout[idx] = (bf16_t)hv;
            out[((size_t)t * BSZ + bb) * (2 * HID) + dir * HID + u] = hv;
            if (s == SEQ - 1) {
                float* fb = finals + (size_t)dir * 2 * BSZ * HID;
                fb[idx] = hv;                         // hT
                fb[(size_t)BSZ * HID + idx] = cv;     // cT
            }
        }

        // ---- Device-wide step barrier (skip after final step) ----
        if (s + 1 < SEQ) {
            __threadfence();      // drain this thread's stores to device scope
            __syncthreads();      // whole block done with step s
            if (threadIdx.x == 0)
                __hip_atomic_fetch_add(barrier_cnt, 1u, __ATOMIC_RELEASE,
                                       __HIP_MEMORY_SCOPE_AGENT);
            const unsigned tgt = (unsigned)NBLOCKS * (unsigned)(s + 1);
            // Every thread spins with an agent-scope acquire load so each
            // CU's L0 is invalidated before re-reading the ping-pong h buffer.
            while (__hip_atomic_load(barrier_cnt, __ATOMIC_ACQUIRE,
                                     __HIP_MEMORY_SCOPE_AGENT) < tgt)
                __builtin_amdgcn_s_sleep(2);
        }
    }
}

// ---------------------------------------------------------------------------
// Host side
// ---------------------------------------------------------------------------
extern "C" void kernel_launch(void* const* d_in, const int* in_sizes, int n_in,
                              void* d_out, int out_size, void* d_ws, size_t ws_size,
                              hipStream_t stream) {
    (void)in_sizes; (void)n_in; (void)out_size; (void)ws_size;

    // Inputs (setup_inputs order)
    const float* x     = (const float*)d_in[0];   // (T,B,I)
    const float* h0_f  = (const float*)d_in[1];
    const float* c0_f  = (const float*)d_in[2];
    const float* h0_b  = (const float*)d_in[3];
    const float* c0_b  = (const float*)d_in[4];
    const float* Wih_f = (const float*)d_in[5];   // (4H,I)
    const float* Whh_f = (const float*)d_in[6];   // (4H,H)
    const float* bih_f = (const float*)d_in[7];
    const float* bhh_f = (const float*)d_in[8];
    const float* Wih_b = (const float*)d_in[9];
    const float* Whh_b = (const float*)d_in[10];
    const float* bih_b = (const float*)d_in[11];
    const float* bhh_b = (const float*)d_in[12];

    // Workspace layout (all chunks are multiples of 256 B)
    char* ws = (char*)d_ws;
    size_t off = 0;
    bf16_t* Xbf = (bf16_t*)(ws + off);  off += (size_t)SEQ * BSZ * INP * sizeof(bf16_t);
    bf16_t* WihBf = (bf16_t*)(ws + off); off += (size_t)2 * G4 * INP * sizeof(bf16_t);
    bf16_t* WhhBf = (bf16_t*)(ws + off); off += (size_t)2 * G4 * HID * sizeof(bf16_t);
    bf16_t* hState = (bf16_t*)(ws + off); off += (size_t)2 * 2 * BSZ * HID * sizeof(bf16_t);
    float* cState = (float*)(ws + off);  off += (size_t)2 * BSZ * HID * sizeof(float);
    float* bsum = (float*)(ws + off);    off += (size_t)2 * G4 * sizeof(float);
    unsigned* barCnt = (unsigned*)(ws + off); off += 256;
    float* Gpre = (float*)(ws + off);    off += (size_t)2 * SEQ * BSZ * G4 * sizeof(float);

    float* out    = (float*)d_out;                     // (T,B,2H)
    float* finals = out + (size_t)SEQ * BSZ * 2 * HID; // hT_f, cT_f, hT_b, cT_b

    // --- Phase 1: precision conversion & prep ------------------------------
    {
        const int thr = 256;
        k_cvt_f32_to_bf16<<<1024, thr, 0, stream>>>(Xbf, x, SEQ * BSZ * INP);
        k_cvt_f32_to_bf16<<<256, thr, 0, stream>>>(WihBf,            Wih_f, G4 * INP);
        k_cvt_f32_to_bf16<<<256, thr, 0, stream>>>(WihBf + (size_t)G4 * INP, Wih_b, G4 * INP);
        k_cvt_f32_to_bf16<<<256, thr, 0, stream>>>(WhhBf,            Whh_f, G4 * HID);
        k_cvt_f32_to_bf16<<<256, thr, 0, stream>>>(WhhBf + (size_t)G4 * HID, Whh_b, G4 * HID);
        // h0 into ping buffer 0 of each direction
        k_cvt_f32_to_bf16<<<128, thr, 0, stream>>>(hState,                         h0_f, BSZ * HID);
        k_cvt_f32_to_bf16<<<128, thr, 0, stream>>>(hState + (size_t)2 * BSZ * HID, h0_b, BSZ * HID);
        // combined biases
        k_bias_sum<<<(G4 + thr - 1) / thr, thr, 0, stream>>>(bsum,      bih_f, bhh_f, G4);
        k_bias_sum<<<(G4 + thr - 1) / thr, thr, 0, stream>>>(bsum + G4, bih_b, bhh_b, G4);
        // c0 (f32) straight into c_state
        hipMemcpyAsync(cState,             c0_f, (size_t)BSZ * HID * sizeof(float),
                       hipMemcpyDeviceToDevice, stream);
        hipMemcpyAsync(cState + BSZ * HID, c0_b, (size_t)BSZ * HID * sizeof(float),
                       hipMemcpyDeviceToDevice, stream);
        // software grid-barrier counter
        k_zero_u32<<<1, 1, 0, stream>>>(barCnt);
    }

    // --- Phase 2: time-parallel input-projection GEMM (WMMA bf16) ----------
    {
        dim3 grid((SEQ * BSZ) / 64, G4 / 64, 2); // (512, 32, 2)
        k_input_gemm<<<grid, 128, 0, stream>>>(Xbf, WihBf, bsum, Gpre);
    }

    // --- Phase 3: persistent recurrence (single launch, grid sync) ---------
    {
        dim3 grid(HID / 16, 2); // (32, 2) = 64 blocks, all co-resident
        k_lstm_persistent<<<grid, 128, 0, stream>>>(Gpre, WhhBf, hState, cState,
                                                    out, finals, barCnt);
    }
}
